// ExpertChoiceMoEMatcher_33947421507659
// MI455X (gfx1250) — compile-verified, matrix operand order checked by reading, not run on hardware
//
#include <hip/hip_runtime.h>
#include <math.h>
#include <stdint.h>

// ---------------- problem constants ----------------
#define BT      65536
#define DD      256
#define NE      64
#define KN      2048
#define K2      512      // 2*D

typedef __bf16 bf16_t;
typedef __attribute__((ext_vector_type(16))) __bf16         v16bf;
typedef __attribute__((ext_vector_type(8)))  float          v8f;
typedef __attribute__((ext_vector_type(4)))  unsigned int   uint4v;
typedef __attribute__((ext_vector_type(8)))  unsigned short ushort8;

__device__ __forceinline__ unsigned short f2bf(float f) {
    unsigned u = __float_as_uint(f);
    u += 0x7FFFu + ((u >> 16) & 1u);     // round-to-nearest-even
    return (unsigned short)(u >> 16);
}

__device__ __forceinline__ unsigned sortkey(float f) {
    unsigned u = __float_as_uint(f);
    return u ^ (((unsigned)((int)u >> 31)) | 0x80000000u);   // monotonic float->uint
}

// async DMA: global -> LDS, 16B per lane, tracked by ASYNCcnt
__device__ __forceinline__ void async_b128(unsigned lds_byte_addr, const void* gaddr) {
    asm volatile("global_load_async_to_lds_b128 %0, %1, off"
                 :: "v"(lds_byte_addr), "v"(gaddr)
                 : "memory");
}

// ---------------- zero fill ----------------
__global__ void zero_kernel(float* __restrict__ p, size_t n) {
    size_t i = (size_t)blockIdx.x * blockDim.x + threadIdx.x;
    size_t stride = (size_t)gridDim.x * blockDim.x;
    for (; i < n; i += stride) p[i] = 0.0f;
}

// ---------------- gating GEMM: scores[t][e] = x_flat[t][:512] . gw[:,e] ----------------
__global__ __launch_bounds__(256) void gate_scores(const float* __restrict__ x,
                                                   const float* __restrict__ gw,
                                                   float* __restrict__ scores) {
    __shared__ float xs[64][65];
    __shared__ float gs[64][64];
    const int tid  = threadIdx.x;
    const int tok0 = blockIdx.x * 64;
    const int e    = tid & 63;
    const int trow = tid >> 6;            // 0..3, constant per wave
    float acc[16];
    #pragma unroll
    for (int i = 0; i < 16; ++i) acc[i] = 0.0f;

    for (int kb = 0; kb < K2; kb += 64) {
        #pragma unroll
        for (int s = 0; s < 16; ++s) {
            int idx = tid + s * 256;
            int r = idx >> 6, c = idx & 63;
            xs[r][c] = x[(size_t)(tok0 + r) * K2 + kb + c];
            gs[r][c] = gw[(size_t)(kb + r) * NE + c];
        }
        __syncthreads();
        for (int kk = 0; kk < 64; ++kk) {
            float g = gs[kk][e];
            #pragma unroll
            for (int i = 0; i < 16; ++i) acc[i] += xs[trow + i * 4][kk] * g;
        }
        __syncthreads();
    }
    #pragma unroll
    for (int i = 0; i < 16; ++i)
        scores[(size_t)(tok0 + trow + i * 4) * NE + e] = acc[i];
}

// ---------------- per-expert histogram over 16-bit prefix keys ----------------
__global__ void hist_kernel(const float* __restrict__ scores, unsigned* __restrict__ hist) {
    size_t i = (size_t)blockIdx.x * blockDim.x + threadIdx.x;   // BT*NE total
    unsigned e = (unsigned)(i & 63);
    unsigned p = sortkey(scores[i]) >> 16;
    atomicAdd(&hist[e * 65536u + p], 1u);
}

// ---------------- find k-th threshold bin per expert ----------------
__global__ __launch_bounds__(256) void thresh_kernel(const unsigned* __restrict__ hist,
                                                     unsigned* __restrict__ thr, int k) {
    __shared__ unsigned csum[256];
    const int e = blockIdx.x;
    const unsigned* h = hist + (size_t)e * 65536;
    const int ti = threadIdx.x;
    unsigned s = 0;
    for (int b = 0; b < 256; ++b) s += h[ti * 256 + b];
    csum[ti] = s;
    __syncthreads();
    if (ti == 0) {
        unsigned acc = 0; int c;
        for (c = 255; c >= 0; --c) { if (acc + csum[c] >= (unsigned)k) break; acc += csum[c]; }
        if (c < 0) c = 0;
        unsigned accb = acc; int b;
        for (b = 255; b >= 0; --b) { unsigned cnt = h[c * 256 + b]; if (accb + cnt >= (unsigned)k) break; accb += cnt; }
        if (b < 0) b = 0;
        thr[e * 4 + 0] = (unsigned)(c * 256 + b);   // threshold prefix
        thr[e * 4 + 1] = accb;                      // strictly above
        thr[e * 4 + 2] = (unsigned)k - accb;        // take from threshold bin
    }
}

// ---------------- compact top-k, emit idx/vals, bump counts ----------------
__global__ void select_kernel(const float* __restrict__ scores, const unsigned* __restrict__ thr,
                              unsigned* __restrict__ cnt, int* __restrict__ tIdx,
                              float* __restrict__ tVals, float* __restrict__ oIdx,
                              float* __restrict__ oVals, float* __restrict__ counts) {
    size_t i = (size_t)blockIdx.x * blockDim.x + threadIdx.x;
    int t = (int)(i >> 6), e = (int)(i & 63);
    float s = scores[i];
    unsigned p = sortkey(s) >> 16;
    unsigned tp = thr[e * 4], ca = thr[e * 4 + 1], take = thr[e * 4 + 2];
    int slot = -1;
    if (p > tp) {
        slot = (int)atomicAdd(&cnt[e * 2], 1u);
    } else if (p == tp) {
        unsigned q = atomicAdd(&cnt[e * 2 + 1], 1u);
        if (q < take) slot = (int)(ca + q);
    }
    if (slot >= 0) {
        tIdx[e * KN + slot]  = t;
        tVals[e * KN + slot] = s;
        oIdx[(size_t)slot * NE + e]  = (float)t;
        oVals[(size_t)slot * NE + e] = s;
        atomicAdd(&counts[t], 1.0f);
    }
}

// ---------------- gather selected tokens into bf16 A = [xr | xi], [E][KN][K2] ----------------
__global__ void gather_kernel(const float* __restrict__ x, const int* __restrict__ tIdx,
                              unsigned short* __restrict__ A) {
    size_t i = (size_t)blockIdx.x * blockDim.x + threadIdx.x;   // E*KN*K2
    int e   = (int)(i >> 20);
    int rem = (int)(i & 0xFFFFF);
    int kk  = rem >> 9, j = rem & 511;
    int t   = tIdx[e * KN + kk];
    float v = (j < DD) ? x[(size_t)t * K2 + 2 * j]
                       : x[(size_t)t * K2 + 2 * (j - DD) + 1];
    A[i] = f2bf(v);
}

// ---------------- build bf16 B^T (N-major): B = [[wr, wi],[-wi, wr]], [E][512 N][512 K] ----------------
__global__ void buildB_kernel(const float* __restrict__ w, unsigned short* __restrict__ Bt) {
    size_t i = (size_t)blockIdx.x * blockDim.x + threadIdx.x;   // E*512*512
    int e   = (int)(i >> 18);
    int rem = (int)(i & 0x3FFFF);
    int n = rem >> 9, kq = rem & 511;
    float v;
    if (kq < DD) {
        int d = kq;
        v = (n < DD) ? w[(((size_t)e * DD + d) * DD + n) * 2 + 0]
                     : w[(((size_t)e * DD + d) * DD + (n - DD)) * 2 + 1];
    } else {
        int d = kq - DD;
        v = (n < DD) ? -w[(((size_t)e * DD + d) * DD + n) * 2 + 1]
                     :  w[(((size_t)e * DD + d) * DD + (n - DD)) * 2 + 0];
    }
    Bt[i] = f2bf(v);
}

// ---------------- WMMA expert GEMM with async double-buffered LDS staging ----------------
// Y[e] = A[e] (2048x512) * B[e] (512x512), weighted scatter-add into out
__global__ __launch_bounds__(256) void expert_gemm(const unsigned short* __restrict__ A,
                                                   const unsigned short* __restrict__ Bt,
                                                   const int* __restrict__ tIdx,
                                                   const float* __restrict__ tVals,
                                                   float* __restrict__ out) {
    __shared__ unsigned short sA[2][128 * 32];   // 2 x 8 KB
    __shared__ unsigned short sB[2][128 * 32];   // 2 x 8 KB
    const int e  = blockIdx.z;
    const int m0 = blockIdx.y * 128;
    const int n0 = blockIdx.x * 128;
    const int tid = threadIdx.x;
    const int lane = tid & 31, wid = tid >> 5;
    const int waveM = wid >> 2, waveN = wid & 3;        // 2 x 4 wave grid
    const unsigned short* Ae = A  + (size_t)e * KN * K2;
    const unsigned short* Be = Bt + (size_t)e * K2 * K2;

    // LDS byte addresses (generic LDS pointer truncates to LDS address, ISA aperture rule)
    const unsigned ldsA0 = (unsigned)(uintptr_t)(void*)&sA[0][0];
    const unsigned ldsB0 = (unsigned)(uintptr_t)(void*)&sB[0][0];

    v8f acc[4][2];
    const v8f vzero = {0.f, 0.f, 0.f, 0.f, 0.f, 0.f, 0.f, 0.f};
    #pragma unroll
    for (int i = 0; i < 4; ++i)
        #pragma unroll
        for (int j = 0; j < 2; ++j) acc[i][j] = vzero;

    const int hi = (lane >= 16) ? 8 : 0;                 // K-run base per ISA A-layout
    const int mLane = lane & 15;

    // issue one K-tile (128x32 halves of A and B) into LDS buffer `buf` via async DMA:
    // 2 chunks of 16B per thread per matrix -> 4 async instructions per wave per tile
    auto issue_tile = [&](int buf, int kb) {
        #pragma unroll
        for (int s = 0; s < 2; ++s) {
            int c = tid + s * 256;                       // 0..511 16B chunks
            int row = c >> 2, sub = c & 3;
            async_b128(ldsA0 + (unsigned)buf * 8192u + (unsigned)c * 16u,
                       Ae + (size_t)(m0 + row) * K2 + kb + sub * 8);
            async_b128(ldsB0 + (unsigned)buf * 8192u + (unsigned)c * 16u,
                       Be + (size_t)(n0 + row) * K2 + kb + sub * 8);
        }
    };

    const int NT = K2 / 32;                              // 16 K-tiles
    issue_tile(0, 0);
    for (int it = 0; it < NT; ++it) {
        const int cur = it & 1;
        if (it + 1 < NT) {
            issue_tile((it + 1) & 1, (it + 1) * 32);     // overlap copy of tile i+1
            asm volatile("s_wait_asynccnt 0x4" ::: "memory");   // tile i resident
        } else {
            asm volatile("s_wait_asynccnt 0x0" ::: "memory");
        }
        __syncthreads();

        if (it + 2 < NT)                                  // -> global_prefetch_b8
            __builtin_prefetch(Ae + (size_t)(m0 + (tid >> 1)) * K2 + (it + 2) * 32, 0, 0);

        v16bf afr[4], bfr[2];
        #pragma unroll
        for (int i = 0; i < 4; ++i) {
            int r = waveM * 64 + i * 16 + mLane;
            const ushort8* p = (const ushort8*)(&sA[cur][0] + r * 32 + hi);
            union { ushort8 u[2]; v16bf v; } tm;
            tm.u[0] = p[0]; tm.u[1] = p[2];              // runs at +0 and +16 halves
            afr[i] = tm.v;
        }
        #pragma unroll
        for (int j = 0; j < 2; ++j) {
            int r = waveN * 32 + j * 16 + mLane;
            const ushort8* p = (const ushort8*)(&sB[cur][0] + r * 32 + hi);
            union { ushort8 u[2]; v16bf v; } tm;
            tm.u[0] = p[0]; tm.u[1] = p[2];
            bfr[j] = tm.v;
        }
        #pragma unroll
        for (int i = 0; i < 4; ++i)
            #pragma unroll
            for (int j = 0; j < 2; ++j)
                acc[i][j] = __builtin_amdgcn_wmma_f32_16x16x32_bf16(
                    false, afr[i], false, bfr[j], (short)0, acc[i][j], false, false);
        __syncthreads();                                 // done reading buf `cur` before reuse
    }

    // weighted scatter-add: C/D layout -> lane 0-15: M=v,N=lane ; lane 16-31: M=8+v,N=lane-16
    const int rHi = (lane >= 16) ? 8 : 0;
    const int nl  = lane & 15;
    #pragma unroll
    for (int i = 0; i < 4; ++i) {
        int baseM = m0 + waveM * 64 + i * 16;
        int   tt[8];
        float ww[8];
        #pragma unroll
        for (int v = 0; v < 8; ++v) {
            int gk = baseM + v + rHi;
            tt[v] = tIdx[e * KN + gk];
            ww[v] = tVals[e * KN + gk];
        }
        #pragma unroll
        for (int j = 0; j < 2; ++j) {
            int gc = n0 + waveN * 32 + j * 16 + nl;
            int d, comp;
            if (gc < DD) { d = gc;      comp = 0; }
            else         { d = gc - DD; comp = 1; }
            #pragma unroll
            for (int v = 0; v < 8; ++v)
                atomicAdd(&out[(size_t)tt[v] * K2 + d * 2 + comp], acc[i][j][v] * ww[v]);
        }
    }
}

// ---------------- epilogue: avg + bias + exact GELU ----------------
__global__ void finalize_kernel(float* __restrict__ res, const float* __restrict__ counts,
                                const float* __restrict__ bias) {
    size_t i = (size_t)blockIdx.x * blockDim.x + threadIdx.x;   // BT*K2
    int t = (int)(i >> 9);
    int j = (int)(i & 511);
    int d = j >> 1;
    float v = res[i] / fmaxf(counts[t], 1.0f) + bias[d];
    res[i] = 0.5f * v * (1.0f + erff(v * 0.70710678118654752f));
}

// ---------------- launch ----------------
extern "C" void kernel_launch(void* const* d_in, const int* in_sizes, int n_in,
                              void* d_out, int out_size, void* d_ws, size_t ws_size,
                              hipStream_t stream) {
    const float* x    = (const float*)d_in[0];
    const float* gw   = (const float*)d_in[1];
    const float* wexp = (const float*)d_in[2];
    const float* bias = (const float*)d_in[3];
    // k_nodes (d_in[4]) fixed at 2048

    float* out = (float*)d_out;
    char*  ws  = (char*)d_ws;

    float*          scores = (float*)(ws + 0);            // 16 MB
    unsigned*       hist   = (unsigned*)(ws + 16777216);   // 16 MB
    unsigned*       thr    = (unsigned*)(ws + 33554432);   // 1 KB
    unsigned*       cnt    = (unsigned*)(ws + 33555456);   // 0.5 KB
    int*            tIdx   = (int*)(ws + 33556480);        // 512 KB
    float*          tVals  = (float*)(ws + 34080768);      // 512 KB
    unsigned short* A      = (unsigned short*)(ws + 34605056);   // 128 MB
    unsigned short* Bt     = (unsigned short*)(ws + 168822784);  // 32 MB

    float* res   = out;                    // BT*512
    float* oIdx  = out + 33554432;         // KN*NE
    float* oVals = out + 33685504;         // KN*NE
    float* cnts  = out + 33816576;         // BT

    zero_kernel<<<2048, 256, 0, stream>>>((float*)d_out, (size_t)33882112);
    zero_kernel<<<2048, 256, 0, stream>>>((float*)hist, (size_t)(NE * 65536));
    zero_kernel<<<1,    256, 0, stream>>>((float*)cnt, (size_t)128);

    gate_scores<<<BT / 64, 256, 0, stream>>>(x, gw, scores);
    hist_kernel<<<(BT * NE) / 256, 256, 0, stream>>>(scores, hist);
    thresh_kernel<<<NE, 256, 0, stream>>>(hist, thr, KN);
    select_kernel<<<(BT * NE) / 256, 256, 0, stream>>>(scores, thr, cnt, tIdx, tVals, oIdx, oVals, cnts);
    gather_kernel<<<(NE * KN * K2) / 256, 256, 0, stream>>>(x, tIdx, A);
    buildB_kernel<<<(NE * K2 * K2) / 256, 256, 0, stream>>>(wexp, Bt);

    dim3 gg(K2 / 128, KN / 128, NE);   // (4, 16, 64)
    expert_gemm<<<gg, 256, 0, stream>>>(A, Bt, tIdx, tVals, res);

    finalize_kernel<<<(BT * K2) / 256, 256, 0, stream>>>(res, cnts, bias);
}
